// AblationAttention_75307956568358
// MI455X (gfx1250) — matrix operand order repaired and checked by reading, not compiled, not run on hardware
//
#include <hip/hip_runtime.h>
#include <hip/hip_bf16.h>

// ---------------------------------------------------------------------------
// Problem constants (from reference): B=2, T=2048, E=1024, H=16, D=64, W=256
// ---------------------------------------------------------------------------
constexpr int BB = 2;
constexpr int T  = 2048;
constexpr int E  = 1024;
constexpr int H  = 16;
constexpr int D  = 64;
constexpr int WIN = 256;
constexpr int M  = BB * T;     // 4096 total rows
constexpr int NK = E / 32;     // 32 k-steps per GEMM
constexpr int NT = E / 16;     // 64 column tiles per weight matrix
constexpr size_t NT_STRIDE = (size_t)NK * 32 * 16;   // elements between ntiles in packed W

typedef __attribute__((ext_vector_type(16))) __bf16 v16bf;
typedef __attribute__((ext_vector_type(8)))  float  v8f;

// ---------------------------------------------------------------------------
// Helpers
// ---------------------------------------------------------------------------
__device__ __forceinline__ __bf16 f2bf(float f) {
    union { float f; unsigned u; } U; U.f = f;
    unsigned u = U.u;
    unsigned r = u + 0x7FFFu + ((u >> 16) & 1u);          // round to nearest even
    if ((u & 0x7F800000u) == 0x7F800000u) r = u;          // inf/nan: truncate
    union { unsigned short s; __bf16 b; } V;
    V.s = (unsigned short)(r >> 16);
    return V.b;
}

__device__ __forceinline__ v8f wmma_bf16(v16bf a, v16bf b, v8f c) {
    return __builtin_amdgcn_wmma_f32_16x16x32_bf16(
        /*neg_a=*/false, a, /*neg_b=*/false, b,
        /*c_mod=*/(short)0, c, /*reuse_a=*/false, /*reuse_b=*/false);
}

// A-frag (16x32 bf16, row-major source): two contiguous 16B chunks per lane.
__device__ __forceinline__ v16bf load_a_frag(const __bf16* base, int lda, int lane) {
    const int half = (lane >> 4) & 1;
    const int r    = lane & 15;
    const __bf16* p = base + (size_t)r * lda + half * 16;
    union { uint4 u[2]; v16bf v; } U;
    U.u[0] = *(const uint4*)(p);
    U.u[1] = *(const uint4*)(p + 8);
    return U.v;
}

// Fragment whose 16 per-lane elements are contiguous in memory (32 bytes).
__device__ __forceinline__ v16bf load_contig16(const __bf16* p) {
    union { uint4 u[2]; v16bf v; } U;
    U.u[0] = *(const uint4*)(p);
    U.u[1] = *(const uint4*)(p + 8);
    return U.v;
}

// Row reductions across the 16 lanes of each half-wave (C/D layout).
__device__ __forceinline__ void row_reduce_sum8(float v[8]) {
#pragma unroll
    for (int m = 1; m <= 8; m <<= 1)
#pragma unroll
        for (int r = 0; r < 8; ++r) v[r] += __shfl_xor(v[r], m, 32);
}
__device__ __forceinline__ void row_reduce_max8(float v[8]) {
#pragma unroll
    for (int m = 1; m <= 8; m <<= 1)
#pragma unroll
        for (int r = 0; r < 8; ++r) v[r] = fmaxf(v[r], __shfl_xor(v[r], m, 32));
}

// ---------------------------------------------------------------------------
// Kernel 0a: f32 -> bf16 elementwise conversion (for x)
// ---------------------------------------------------------------------------
__global__ void cvt_f32_bf16_kernel(const float* __restrict__ src,
                                    __bf16* __restrict__ dst, int n) {
    int i = blockIdx.x * blockDim.x + threadIdx.x;
    if (i < n) dst[i] = f2bf(src[i]);
}

// ---------------------------------------------------------------------------
// Kernel 0b: convert + pack weight matrix [E,E] f32 (row-major, K x N) into
// WMMA B-fragment order: packed[ntile][kstep][lane][16], so a wave's fragment
// load is one contiguous, fully-coalesced 1KB stretch (2 x b128 per lane).
// ---------------------------------------------------------------------------
__global__ void pack_w_kernel(const float* __restrict__ src, __bf16* __restrict__ dst) {
    int tid   = blockIdx.x * blockDim.x + threadIdx.x;  // NT*NK*32 = 65536 threads
    int lane  = tid & 31;
    int ks    = (tid >> 5) & (NK - 1);
    int ntile = tid >> 10;
    int half  = lane >> 4, lh = lane & 15;
    const float* p = src + (size_t)(ks * 32 + 16 * half) * E + ntile * 16 + lh;
    union { uint4 u[2]; __bf16 h16[16]; } U;
#pragma unroll
    for (int e = 0; e < 16; ++e) U.h16[e] = f2bf(p[(size_t)e * E]);
    uint4* q = (uint4*)(dst + ((size_t)(ntile * NK + ks) * 32 + lane) * 16);
    q[0] = U.u[0];
    q[1] = U.u[1];
}

// ---------------------------------------------------------------------------
// Kernel 1: fused QKV projection + per-head RMSNorm.
// Each wave: 32 rows x one head's 64 columns (2 A-frags x 4 B-frags = 8 WMMAs
// per k-step; B-fragments reused across both row tiles). z = {0,1,2} = q,k,v.
// Q,K stored [B,H,T,D] bf16; V stored transposed [B,H,D,T] bf16.
// ---------------------------------------------------------------------------
__global__ void qkv_proj_kernel(const __bf16* __restrict__ xb,
                                const __bf16* __restrict__ wpq,
                                const __bf16* __restrict__ wpk,
                                const __bf16* __restrict__ wpv,
                                const float* __restrict__ bq,
                                const float* __restrict__ bk,
                                const float* __restrict__ bv,
                                const float* __restrict__ qn_w,
                                const float* __restrict__ kn_w,
                                __bf16* __restrict__ qh,
                                __bf16* __restrict__ kh,
                                __bf16* __restrict__ vt) {
    const int lane = threadIdx.x & 31;
    const int wave = threadIdx.x >> 5;
    const int half = lane >> 4, lh = lane & 15;
    const int m0   = (blockIdx.x * 4 + wave) * 32;
    const int h    = blockIdx.y;
    const int which = blockIdx.z;
    const __bf16* Wp   = which == 0 ? wpq : (which == 1 ? wpk : wpv);
    const float*  bias = which == 0 ? bq  : (which == 1 ? bk  : bv);
    const int n0 = h * D;
    const int nt0 = h * 4;   // first 16-col tile of this head

    v8f acc[2][4] = {};
    for (int k0 = 0; k0 < E; k0 += 32) {
        const int ks = k0 >> 5;
        // issue all fragment loads first so WMMAs can overlap remaining loads
        v16bf a0 = load_a_frag(xb + (size_t)m0 * E + k0, E, lane);
        v16bf a1 = load_a_frag(xb + (size_t)(m0 + 16) * E + k0, E, lane);
        const __bf16* bbase = Wp + ((size_t)(nt0 * NK + ks) * 32 + lane) * 16;
        v16bf b0 = load_contig16(bbase);
        v16bf b1 = load_contig16(bbase + NT_STRIDE);
        v16bf b2 = load_contig16(bbase + 2 * NT_STRIDE);
        v16bf b3 = load_contig16(bbase + 3 * NT_STRIDE);
        if (k0 + 32 < E) {
            __builtin_prefetch(xb + (size_t)m0 * E + k0 + 32, 0, 1);
            __builtin_prefetch(bbase + 512, 0, 1);   // next kstep of first ntile
        }
        acc[0][0] = wmma_bf16(a0, b0, acc[0][0]);
        acc[0][1] = wmma_bf16(a0, b1, acc[0][1]);
        acc[0][2] = wmma_bf16(a0, b2, acc[0][2]);
        acc[0][3] = wmma_bf16(a0, b3, acc[0][3]);
        acc[1][0] = wmma_bf16(a1, b0, acc[1][0]);
        acc[1][1] = wmma_bf16(a1, b1, acc[1][1]);
        acc[1][2] = wmma_bf16(a1, b2, acc[1][2]);
        acc[1][3] = wmma_bf16(a1, b3, acc[1][3]);
    }

#pragma unroll
    for (int mi = 0; mi < 2; ++mi) {
        float val[4][8];
#pragma unroll
        for (int t = 0; t < 4; ++t) {
            float bi = bias[n0 + t * 16 + lh];
#pragma unroll
            for (int r = 0; r < 8; ++r) val[t][r] = acc[mi][t][r] + bi;
        }

        if (which < 2) {  // RMSNorm over head dim (the wave's 64 columns)
            const float* gw = which == 0 ? qn_w : kn_w;
            float ss[8];
#pragma unroll
            for (int r = 0; r < 8; ++r)
                ss[r] = val[0][r]*val[0][r] + val[1][r]*val[1][r]
                      + val[2][r]*val[2][r] + val[3][r]*val[3][r];
            row_reduce_sum8(ss);
            float sc[8];
#pragma unroll
            for (int r = 0; r < 8; ++r)
                sc[r] = rsqrtf(ss[r] * (1.0f / (float)D) + 1.1920929e-07f);
#pragma unroll
            for (int t = 0; t < 4; ++t) {
                float g = gw[t * 16 + lh];
#pragma unroll
                for (int r = 0; r < 8; ++r) val[t][r] *= sc[r] * g;
            }
            __bf16* dst = which == 0 ? qh : kh;
#pragma unroll
            for (int t = 0; t < 4; ++t)
#pragma unroll
                for (int r = 0; r < 8; ++r) {
                    int m  = m0 + 16 * mi + r + 8 * half;
                    int bb = m >> 11;          // / T
                    int tt = m & (T - 1);
                    dst[(((size_t)bb * H + h) * T + tt) * D + t * 16 + lh] = f2bf(val[t][r]);
                }
        } else {
            // V transposed [B,H,D,T]: lane's 8 rows = 8 consecutive T slots.
            const int mb = m0 + 16 * mi + 8 * half;
            const int bb = mb >> 11;
            const int tt = mb & (T - 1);
#pragma unroll
            for (int t = 0; t < 4; ++t) {
                union { uint4 u; __bf16 h8[8]; } Upk;
#pragma unroll
                for (int r = 0; r < 8; ++r) Upk.h8[r] = f2bf(val[t][r]);
                size_t idx = (((size_t)bb * H + h) * D + t * 16 + lh) * (size_t)T + tt;
                *(uint4*)(vt + idx) = Upk.u;
            }
        }
    }
}

// ---------------------------------------------------------------------------
// Kernel 2: sliding-window causal flash attention.
// One wave per 16-query tile of one (b,h). Processes 32-key pair-tiles.
// (The reference's fitness term adds a per-row constant before softmax -> no-op.)
// ---------------------------------------------------------------------------
__global__ void attn_kernel(const __bf16* __restrict__ qh,
                            const __bf16* __restrict__ kh,
                            const __bf16* __restrict__ vt,
                            __bf16* __restrict__ ab) {
    __shared__ __align__(32) __bf16 pstage[16 * 32];
    const int lane = threadIdx.x & 31;
    const int half = lane >> 4, lh = lane & 15;
    const int qs = blockIdx.x * 16;
    const int h  = blockIdx.y;
    const int b  = blockIdx.z;
    const size_t bhT = ((size_t)b * H + h) * (size_t)T;
    const size_t bhD = ((size_t)b * H + h) * (size_t)D;

    const __bf16* Qb = qh + (bhT + qs) * D;
    v16bf qa0 = load_a_frag(Qb, D, lane);        // K-dim 0..31
    v16bf qa1 = load_a_frag(Qb + 32, D, lane);   // K-dim 32..63

    v8f o[4] = {};
    float l[8] = {};
    float mrow[8];
#pragma unroll
    for (int r = 0; r < 8; ++r) mrow[r] = -1e30f;

    int klo = qs - (WIN - 1); if (klo < 0) klo = 0;
    for (int kp = klo & ~31; kp <= qs + 15; kp += 32) {
        // ---- S = Q K^T for 16 queries x 32 keys (4 WMMAs) ----
        const __bf16* K0 = kh + (bhT + kp + lh) * D + 16 * half;
        const __bf16* K1 = kh + (bhT + kp + 16 + lh) * D + 16 * half;
        v16bf kb00 = load_contig16(K0);
        v16bf kb01 = load_contig16(K0 + 32);
        v16bf kb10 = load_contig16(K1);
        v16bf kb11 = load_contig16(K1 + 32);
        v8f s0 = {}, s1 = {};
        s0 = wmma_bf16(qa0, kb00, s0);
        s0 = wmma_bf16(qa1, kb01, s0);
        s1 = wmma_bf16(qa0, kb10, s1);
        s1 = wmma_bf16(qa1, kb11, s1);

        // ---- scale + causal/sliding-window mask ----
        float sv[2][8];
#pragma unroll
        for (int kt = 0; kt < 2; ++kt)
#pragma unroll
            for (int r = 0; r < 8; ++r) {
                float x = (kt ? s1[r] : s0[r]) * 0.125f;   // 1/sqrt(64)
                int qg = qs + r + 8 * half;
                int kg = kp + 16 * kt + lh;
                bool bad = (kg > qg) || (qg - kg >= WIN);
                sv[kt][r] = bad ? -__builtin_inff() : x;
            }

        // ---- online softmax ----
        float tm[8];
#pragma unroll
        for (int r = 0; r < 8; ++r) tm[r] = fmaxf(sv[0][r], sv[1][r]);
        row_reduce_max8(tm);
        float sc[8];
#pragma unroll
        for (int r = 0; r < 8; ++r) {
            float mn = fmaxf(mrow[r], fmaxf(tm[r], -1e30f));  // floor guards all-masked tiles
            sc[r] = __expf(mrow[r] - mn);
            mrow[r] = mn;
        }
        float pv[2][8], ps[8] = {};
#pragma unroll
        for (int kt = 0; kt < 2; ++kt)
#pragma unroll
            for (int r = 0; r < 8; ++r) {
                pv[kt][r] = __expf(sv[kt][r] - mrow[r]);
                ps[r] += pv[kt][r];
            }
        row_reduce_sum8(ps);
#pragma unroll
        for (int r = 0; r < 8; ++r) l[r] = l[r] * sc[r] + ps[r];
#pragma unroll
        for (int t = 0; t < 4; ++t)
#pragma unroll
            for (int r = 0; r < 8; ++r) o[t][r] *= sc[r];

        // ---- P: C-layout -> A-layout via LDS ----
#pragma unroll
        for (int kt = 0; kt < 2; ++kt)
#pragma unroll
            for (int r = 0; r < 8; ++r)
                pstage[(r + 8 * half) * 32 + 16 * kt + lh] = f2bf(pv[kt][r]);
        __syncthreads();
        v16bf pa = load_a_frag(pstage, 32, lane);

        // ---- O += P V (4 WMMAs); V^T gives contiguous per-lane fragments ----
        const __bf16* vb0 = vt + (bhD + lh) * (size_t)T + kp + 16 * half;
        v16bf v0 = load_contig16(vb0);
        v16bf v1 = load_contig16(vb0 + 16 * (size_t)T);
        v16bf v2 = load_contig16(vb0 + 32 * (size_t)T);
        v16bf v3 = load_contig16(vb0 + 48 * (size_t)T);
        o[0] = wmma_bf16(pa, v0, o[0]);
        o[1] = wmma_bf16(pa, v1, o[1]);
        o[2] = wmma_bf16(pa, v2, o[2]);
        o[3] = wmma_bf16(pa, v3, o[3]);
        __syncthreads();
    }

    // ---- epilogue: O / l, store bf16 [B,T,E] (head-concat layout) ----
#pragma unroll
    for (int t = 0; t < 4; ++t)
#pragma unroll
        for (int r = 0; r < 8; ++r) {
            int qg = qs + r + 8 * half;
            ab[((size_t)b * T + qg) * E + h * D + t * 16 + lh] =
                f2bf(o[t][r] * (1.0f / l[r]));
        }
}

// ---------------------------------------------------------------------------
// Kernel 3: output projection: out = attn @ wo + bo (f32 output)
// Each wave: 32 rows x 64 columns (8 WMMAs per k-step).
// ---------------------------------------------------------------------------
__global__ void out_proj_kernel(const __bf16* __restrict__ ab,
                                const __bf16* __restrict__ wpo,
                                const float* __restrict__ bo,
                                float* __restrict__ out) {
    const int lane = threadIdx.x & 31;
    const int wave = threadIdx.x >> 5;
    const int half = lane >> 4, lh = lane & 15;
    const int m0 = (blockIdx.x * 4 + wave) * 32;
    const int n0 = blockIdx.y * 64;
    const int nt0 = blockIdx.y * 4;

    v8f acc[2][4] = {};
    for (int k0 = 0; k0 < E; k0 += 32) {
        const int ks = k0 >> 5;
        v16bf a0 = load_a_frag(ab + (size_t)m0 * E + k0, E, lane);
        v16bf a1 = load_a_frag(ab + (size_t)(m0 + 16) * E + k0, E, lane);
        const __bf16* bbase = wpo + ((size_t)(nt0 * NK + ks) * 32 + lane) * 16;
        v16bf b0 = load_contig16(bbase);
        v16bf b1 = load_contig16(bbase + NT_STRIDE);
        v16bf b2 = load_contig16(bbase + 2 * NT_STRIDE);
        v16bf b3 = load_contig16(bbase + 3 * NT_STRIDE);
        if (k0 + 32 < E) {
            __builtin_prefetch(ab + (size_t)m0 * E + k0 + 32, 0, 1);
            __builtin_prefetch(bbase + 512, 0, 1);
        }
        acc[0][0] = wmma_bf16(a0, b0, acc[0][0]);
        acc[0][1] = wmma_bf16(a0, b1, acc[0][1]);
        acc[0][2] = wmma_bf16(a0, b2, acc[0][2]);
        acc[0][3] = wmma_bf16(a0, b3, acc[0][3]);
        acc[1][0] = wmma_bf16(a1, b0, acc[1][0]);
        acc[1][1] = wmma_bf16(a1, b1, acc[1][1]);
        acc[1][2] = wmma_bf16(a1, b2, acc[1][2]);
        acc[1][3] = wmma_bf16(a1, b3, acc[1][3]);
    }
#pragma unroll
    for (int mi = 0; mi < 2; ++mi)
#pragma unroll
        for (int t = 0; t < 4; ++t) {
            float bi = bo[n0 + t * 16 + lh];
#pragma unroll
            for (int r = 0; r < 8; ++r) {
                int m = m0 + 16 * mi + r + 8 * half;
                out[(size_t)m * E + n0 + t * 16 + lh] = acc[mi][t][r] + bi;
            }
        }
}

// ---------------------------------------------------------------------------
// Launch
// ---------------------------------------------------------------------------
extern "C" void kernel_launch(void* const* d_in, const int* in_sizes, int n_in,
                              void* d_out, int out_size, void* d_ws, size_t ws_size,
                              hipStream_t stream) {
    const float* x  = (const float*)d_in[0];
    const float* wq = (const float*)d_in[1];
    const float* bq = (const float*)d_in[2];
    const float* wk = (const float*)d_in[3];
    const float* bk = (const float*)d_in[4];
    const float* wv = (const float*)d_in[5];
    const float* bv = (const float*)d_in[6];
    const float* wo = (const float*)d_in[7];
    const float* bo = (const float*)d_in[8];
    const float* qn = (const float*)d_in[9];
    const float* kn = (const float*)d_in[10];
    float* out = (float*)d_out;

    char* ws = (char*)d_ws;
    constexpr size_t MiB = 1u << 20;
    __bf16* xb   = (__bf16*)(ws + 0);          // 8 MiB: x bf16 [M,E]
    __bf16* wpq  = (__bf16*)(ws + 8  * MiB);   // 2 MiB each, packed frag layout
    __bf16* wpk  = (__bf16*)(ws + 10 * MiB);
    __bf16* wpv  = (__bf16*)(ws + 12 * MiB);
    __bf16* wpo  = (__bf16*)(ws + 14 * MiB);
    __bf16* qh   = (__bf16*)(ws + 16 * MiB);   // 8 MiB, [B,H,T,D]
    __bf16* kh   = (__bf16*)(ws + 24 * MiB);   // 8 MiB, [B,H,T,D]
    __bf16* vt   = (__bf16*)(ws + 32 * MiB);   // 8 MiB, [B,H,D,T] (transposed)
    __bf16* abuf = (__bf16*)(ws + 40 * MiB);   // 8 MiB, attn out bf16 [B,T,E]

    // 0) convert x; convert+pack the 4 weight matrices into fragment order
    {
        int n = M * E;
        cvt_f32_bf16_kernel<<<(n + 255) / 256, 256, 0, stream>>>(x, xb, n);
        int pthreads = NT * NK * 32;  // 65536
        pack_w_kernel<<<pthreads / 256, 256, 0, stream>>>(wq, wpq);
        pack_w_kernel<<<pthreads / 256, 256, 0, stream>>>(wk, wpk);
        pack_w_kernel<<<pthreads / 256, 256, 0, stream>>>(wv, wpv);
        pack_w_kernel<<<pthreads / 256, 256, 0, stream>>>(wo, wpo);
    }

    // 1) fused QKV projection + RMSNorm (each wave: 32 rows x 64 cols)
    qkv_proj_kernel<<<dim3(M / 128, H, 3), 128, 0, stream>>>(
        xb, wpq, wpk, wpv, bq, bk, bv, qn, kn, qh, kh, vt);

    // 2) sliding-window flash attention
    attn_kernel<<<dim3(T / 16, H, BB), 32, 0, stream>>>(qh, kh, vt, abuf);

    // 3) output projection (each wave: 32 rows x 64 cols)
    out_proj_kernel<<<dim3(M / 128, E / 64), 128, 0, stream>>>(abuf, wpo, bo, out);
}
